// Transformer_44676249813142
// MI455X (gfx1250) — matrix operand order repaired and checked by reading, not compiled, not run on hardware
//
#include <hip/hip_runtime.h>
#include <hip/hip_bf16.h>
#include <cstdint>
#include <cstddef>

#define DEV static __device__ __forceinline__

typedef __bf16 bf16_t;
typedef bf16_t v16bf __attribute__((ext_vector_type(16)));
typedef float  v8f   __attribute__((ext_vector_type(8)));

constexpr int B_ = 2, L_ = 2048, D_ = 1024, H_ = 16, DH_ = 64, FF_ = 4096, NL_ = 8;
constexpr int MROWS = B_ * L_;  // 4096

// float -> bf16, round to nearest even
DEV bf16_t f2bf(float f) {
  unsigned u = __float_as_uint(f);
  unsigned r = (u + 0x7fffu + ((u >> 16) & 1u)) >> 16;
  unsigned short s = (unsigned short)r;
  return __builtin_bit_cast(bf16_t, s);
}

// Generic pointer to __shared__ is {shared_aperture_hi, lds_offset}; low 32
// bits are the wave-relative LDS byte address used by DS/async instructions.
DEV unsigned ldsoff(const void* p) { return (unsigned)(unsigned long long)p; }

// Async global->LDS copy, 16 bytes per lane (GVS mode: saddr base + vgpr byte
// offset). Tracked by ASYNCcnt; pair with wait_asynccnt0() + __syncthreads().
DEV void async_copy_b128(void* lds_dst, const void* gbase, unsigned goff) {
  asm volatile("global_load_async_to_lds_b128 %0, %1, %2"
               :
               : "v"(ldsoff(lds_dst)), "v"(goff), "s"(gbase)
               : "memory");
}
DEV void wait_asynccnt0() { asm volatile("s_wait_asynccnt 0x0" ::: "memory"); }

// A-matrix 16x32 bf16 fragment (row-major source, 32 contiguous K per row):
// lane m = lane%16, g = lane>>4 holds K in {g*8..g*8+7} and {16+g*8..16+g*8+7}
DEV v16bf load_a_frag(const bf16_t* row, int g) {
  union { v16bf v; uint4 u[2]; } r;
  r.u[0] = *(const uint4*)(row + g * 8);
  r.u[1] = *(const uint4*)(row + 16 + g * 8);
  return r.v;
}

// B-matrix 32x16 bf16 fragment from B^T stored row-major (col-major B):
// lane n = lane%16, g = lane>>4 holds K in {g*8..g*8+7} and {16+g*8..16+g*8+7}
DEV v16bf load_b_frag(const bf16_t* col, int g) {
  union { v16bf v; uint4 u[2]; } r;
  r.u[0] = *(const uint4*)(col + g * 8);
  r.u[1] = *(const uint4*)(col + 16 + g * 8);
  return r.v;
}

DEV v8f wmma_bf16(v16bf a, v16bf b, v8f c) {
  return __builtin_amdgcn_wmma_f32_16x16x32_bf16(false, a, false, b, (short)0, c, false, false);
}

// ---------------------------------------------------------------- add PE
__global__ __launch_bounds__(256) void add_pe_kernel(const float* __restrict__ x,
                                                     const float* __restrict__ pe,
                                                     float* __restrict__ h,
                                                     bf16_t* __restrict__ hb) {
  size_t i = (size_t)blockIdx.x * 256 + threadIdx.x;
  float v = x[i] + pe[i % ((size_t)L_ * D_)];
  h[i] = v;
  hb[i] = f2bf(v);
}

// ---------------------------------------------------------------- GEMM
// C[M=4096, N] = A_bf16[M,K] @ W_f32[K,N]   (W converted to bf16 on LDS stage)
// MODE 0: out bf16 reshaped [B,H,L,DH] (QKV), no bias
// MODE 1: out bf16 row-major, +bias, SELU  (MLP up-proj)
// MODE 2: out f32 row-major, +bias          (MLP down-proj)
template <int MODE>
__global__ __launch_bounds__(256) void gemm_kernel(const bf16_t* __restrict__ A,
                                                   const float* __restrict__ W,
                                                   const float* __restrict__ bias,
                                                   void* __restrict__ outp,
                                                   int N, int K) {
  __shared__ bf16_t As[128][32];   // 8 KB
  __shared__ bf16_t Bt[128][32];   // 8 KB (W tile transposed: [n][k])
  const int tid = threadIdx.x;
  const int m0 = blockIdx.y * 128;
  const int n0 = blockIdx.x * 128;
  const int wave = tid >> 5, lane = tid & 31, ln = lane & 15, g = lane >> 4;
  const int wr = wave >> 1, wc = wave & 1;  // 4x2 wave grid: 32 rows x 64 cols each

  v8f zero = {0.f, 0.f, 0.f, 0.f, 0.f, 0.f, 0.f, 0.f};
  v8f acc[2][4];
#pragma unroll
  for (int i = 0; i < 2; i++)
#pragma unroll
    for (int j = 0; j < 4; j++) acc[i][j] = zero;

  const int lrow = tid >> 1, lhalf = tid & 1;      // A loader: 128 rows x 2 halves
  const int wkk = tid >> 3, wn0 = (tid & 7) * 16;  // W loader: 32 k x 8 n-chunks

  for (int kt = 0; kt < K; kt += 32) {
    __syncthreads();
    {  // A tile: two async 16B global->LDS copies per thread (ASYNCcnt path)
      const unsigned abase = (unsigned)(((m0 + lrow) * (size_t)K + kt + lhalf * 16) * 2u);
      async_copy_b128(&As[lrow][lhalf * 16], A, abase);
      async_copy_b128(&As[lrow][lhalf * 16 + 8], A, abase + 16u);
    }
    {  // W tile: f32 -> bf16, transposed into Bt[n][k]; overlaps with asyncs
      const float* w = W + (size_t)(kt + wkk) * N + n0 + wn0;
      if (kt + 32 < K) __builtin_prefetch(w + (size_t)32 * N, 0, 1);
#pragma unroll
      for (int i = 0; i < 16; i += 4) {
        float4 f = *(const float4*)(w + i);
        Bt[wn0 + i + 0][wkk] = f2bf(f.x);
        Bt[wn0 + i + 1][wkk] = f2bf(f.y);
        Bt[wn0 + i + 2][wkk] = f2bf(f.z);
        Bt[wn0 + i + 3][wkk] = f2bf(f.w);
      }
    }
    wait_asynccnt0();
    __syncthreads();
    v16bf af[2], bfr[4];
#pragma unroll
    for (int mt = 0; mt < 2; mt++) af[mt] = load_a_frag(&As[wr * 32 + mt * 16 + ln][0], g);
#pragma unroll
    for (int nt = 0; nt < 4; nt++) bfr[nt] = load_b_frag(&Bt[wc * 64 + nt * 16 + ln][0], g);
#pragma unroll
    for (int mt = 0; mt < 2; mt++)
#pragma unroll
      for (int nt = 0; nt < 4; nt++) acc[mt][nt] = wmma_bf16(af[mt], bfr[nt], acc[mt][nt]);
  }

  // epilogue: C lane layout -> col n = lane%16, rows r + 8*(lane>>4)
#pragma unroll
  for (int mt = 0; mt < 2; mt++) {
#pragma unroll
    for (int nt = 0; nt < 4; nt++) {
      const int n = n0 + wc * 64 + nt * 16 + ln;
      const float bv = (MODE != 0) ? bias[n] : 0.0f;
#pragma unroll
      for (int r = 0; r < 8; r++) {
        const int m = m0 + wr * 32 + mt * 16 + r + 8 * g;
        float val = acc[mt][nt][r] + bv;
        if (MODE == 0) {
          const int bb = m / L_, lp = m % L_, hd = n / DH_, dh = n % DH_;
          ((bf16_t*)outp)[(((size_t)bb * H_ + hd) * L_ + lp) * DH_ + dh] = f2bf(val);
        } else if (MODE == 1) {
          const float sscale = 1.0507009873554805f, salpha = 1.6732632423543772f;
          float sv = val > 0.0f ? sscale * val : sscale * salpha * (__expf(val) - 1.0f);
          ((bf16_t*)outp)[(size_t)m * N + n] = f2bf(sv);
        } else {
          ((float*)outp)[(size_t)m * N + n] = val;
        }
      }
    }
  }
}

// ---------------------------------------------------------------- flash attention
// q,k,v: bf16 [B,H,L,DH]; o: f32 [B,L,D]. Causal, scale = 1/sqrt(DH).
__global__ __launch_bounds__(128) void attn_kernel(const bf16_t* __restrict__ q,
                                                   const bf16_t* __restrict__ k,
                                                   const bf16_t* __restrict__ v,
                                                   float* __restrict__ o) {
  __shared__ bf16_t Ks[32][64];      // key block, row-major (== B^T for Q@K^T)
  __shared__ bf16_t Vt[64][32];      // V block transposed (== B^T for P@V)
  __shared__ bf16_t Pl[4][16][32];   // per-wave P staging
  const int tid = threadIdx.x;
  const int wave = tid >> 5, lane = tid & 31, ln = lane & 15, g = lane >> 4;
  const int bh = blockIdx.y;  // b*H + head
  const int bb = bh / H_, head = bh % H_;
  const int q0wg = blockIdx.x * 64;
  const int q0 = q0wg + wave * 16;
  const float scale = 0.125f;  // 1/sqrt(64)
  const float NEGINF = -__builtin_inff();

  const bf16_t* qp = q + (size_t)bh * L_ * DH_;
  const bf16_t* kp = k + (size_t)bh * L_ * DH_;
  const bf16_t* vp = v + (size_t)bh * L_ * DH_;

  v16bf qf[2];
#pragma unroll
  for (int s = 0; s < 2; s++) qf[s] = load_a_frag(qp + (size_t)(q0 + ln) * DH_ + s * 32, g);

  v8f zero = {0.f, 0.f, 0.f, 0.f, 0.f, 0.f, 0.f, 0.f};
  v8f oacc[4] = {zero, zero, zero, zero};
  float mrow[8], lrow[8];
#pragma unroll
  for (int r = 0; r < 8; r++) { mrow[r] = NEGINF; lrow[r] = 0.0f; }

  const int nkb = blockIdx.x * 2 + 2;  // 32-key blocks covering keys <= q0wg+63
  const int krr = tid >> 2, kc0 = (tid & 3) * 16;
  for (int kbi = 0; kbi < nkb; kbi++) {
    const int kb = kbi * 32;
    __syncthreads();
    {  // K block: async global->LDS (straight copy); V: manual transpose
      const unsigned kbyte = (unsigned)(((kb + krr) * DH_ + kc0) * 2u);
      async_copy_b128(&Ks[krr][kc0], kp, kbyte);
      async_copy_b128(&Ks[krr][kc0 + 8], kp, kbyte + 16u);
      const bf16_t* vs = vp + (size_t)(kb + krr) * DH_ + kc0;
#pragma unroll
      for (int i = 0; i < 16; i++) Vt[kc0 + i][krr] = vs[i];
    }
    wait_asynccnt0();
    __syncthreads();
    // S = Q @ K^T for 2 key sub-tiles, K-dim = DH in 2 steps of 32
    v8f sc[2] = {zero, zero};
#pragma unroll
    for (int f = 0; f < 2; f++)
#pragma unroll
      for (int s = 0; s < 2; s++)
        sc[f] = wmma_bf16(qf[s], load_b_frag(&Ks[f * 16 + ln][s * 32], g), sc[f]);
    // causal mask + online softmax (rows r+8g, this lane owns column ln / 16+ln)
#pragma unroll
    for (int r = 0; r < 8; r++) {
      const int qa = q0 + r + 8 * g;
      float s0 = sc[0][r], s1 = sc[1][r];
      if (kb + ln > qa) s0 = NEGINF;
      if (kb + 16 + ln > qa) s1 = NEGINF;
      float mx = fmaxf(s0, s1);
#pragma unroll
      for (int off = 1; off < 16; off <<= 1) mx = fmaxf(mx, __shfl_xor(mx, off, 32));
      const float mn = fmaxf(mrow[r], mx);
      const float c = __expf((mrow[r] - mn) * scale);
      const float p0 = __expf((s0 - mn) * scale);
      const float p1 = __expf((s1 - mn) * scale);
      float ps = p0 + p1;
#pragma unroll
      for (int off = 1; off < 16; off <<= 1) ps += __shfl_xor(ps, off, 32);
      lrow[r] = lrow[r] * c + ps;
      mrow[r] = mn;
#pragma unroll
      for (int f = 0; f < 4; f++) oacc[f][r] *= c;
      Pl[wave][r + 8 * g][ln] = f2bf(p0);
      Pl[wave][r + 8 * g][16 + ln] = f2bf(p1);
    }
    __syncthreads();
    // O += P @ V  (A = P 16x32, B = V 32x16 per dh tile)
    v16bf pf = load_a_frag(&Pl[wave][ln][0], g);
#pragma unroll
    for (int f = 0; f < 4; f++)
      oacc[f] = wmma_bf16(pf, load_b_frag(&Vt[f * 16 + ln][0], g), oacc[f]);
  }
  // write O / l  into [B, L, D]
#pragma unroll
  for (int f = 0; f < 4; f++) {
    const int dh = f * 16 + ln;
#pragma unroll
    for (int r = 0; r < 8; r++) {
      const int qa = q0 + r + 8 * g;
      o[((size_t)bb * L_ + qa) * D_ + head * DH_ + dh] = oacc[f][r] / lrow[r];
    }
  }
}

// ---------------------------------------------------------------- residual + layernorm
__global__ __launch_bounds__(256) void ln_kernel(float* __restrict__ h,
                                                 const float* __restrict__ delta,
                                                 const float* __restrict__ gmm,
                                                 const float* __restrict__ bta,
                                                 bf16_t* __restrict__ hb) {
  const int tid = threadIdx.x;
  const size_t base = (size_t)blockIdx.x * D_;
  float4 hv = *(const float4*)(h + base + tid * 4);
  float4 dv = *(const float4*)(delta + base + tid * 4);
  float y[4] = {hv.x + dv.x, hv.y + dv.y, hv.z + dv.z, hv.w + dv.w};
  float sum = y[0] + y[1] + y[2] + y[3];
  float sq = y[0] * y[0] + y[1] * y[1] + y[2] * y[2] + y[3] * y[3];
#pragma unroll
  for (int off = 1; off < 32; off <<= 1) {
    sum += __shfl_xor(sum, off, 32);
    sq += __shfl_xor(sq, off, 32);
  }
  __shared__ float s_sum[8], s_sq[8], s_mv[2];
  const int wave = tid >> 5, lane = tid & 31;
  if (lane == 0) { s_sum[wave] = sum; s_sq[wave] = sq; }
  __syncthreads();
  if (tid == 0) {
    float ts = 0.f, tq = 0.f;
    for (int i = 0; i < 8; i++) { ts += s_sum[i]; tq += s_sq[i]; }
    float mean = ts / D_;
    float var = tq / D_ - mean * mean;
    s_mv[0] = mean;
    s_mv[1] = rsqrtf(var + 1e-5f);
  }
  __syncthreads();
  const float mean = s_mv[0], inv = s_mv[1];
#pragma unroll
  for (int i = 0; i < 4; i++) {
    const int c = tid * 4 + i;
    float ov = (y[i] - mean) * inv * gmm[c] + bta[c];
    h[base + c] = ov;
    hb[base + c] = f2bf(ov);
  }
}

// ---------------------------------------------------------------- launch
extern "C" void kernel_launch(void* const* d_in, const int* in_sizes, int n_in,
                              void* d_out, int out_size, void* d_ws, size_t ws_size,
                              hipStream_t stream) {
  (void)in_sizes; (void)n_in; (void)out_size; (void)ws_size;
  const float* x    = (const float*)d_in[0];
  const float* pe   = (const float*)d_in[1];
  const float* wq   = (const float*)d_in[2];
  const float* wk   = (const float*)d_in[3];
  const float* wv   = (const float*)d_in[4];
  const float* ln1g = (const float*)d_in[5];
  const float* ln1b = (const float*)d_in[6];
  const float* w1   = (const float*)d_in[7];
  const float* b1   = (const float*)d_in[8];
  const float* w2   = (const float*)d_in[9];
  const float* b2   = (const float*)d_in[10];
  const float* ln2g = (const float*)d_in[11];
  const float* ln2b = (const float*)d_in[12];

  // workspace carve (~96 MB total)
  char* ws = (char*)d_ws;
  const size_t nBLD = (size_t)B_ * L_ * D_;
  float*  h     = (float*)ws;   ws += nBLD * 4;
  float*  delta = (float*)ws;   ws += nBLD * 4;
  bf16_t* hb    = (bf16_t*)ws;  ws += nBLD * 2;
  bf16_t* qb    = (bf16_t*)ws;  ws += nBLD * 2;
  bf16_t* kb    = (bf16_t*)ws;  ws += nBLD * 2;
  bf16_t* vb    = (bf16_t*)ws;  ws += nBLD * 2;
  bf16_t* ff1   = (bf16_t*)ws;  ws += (size_t)B_ * L_ * FF_ * 2;

  add_pe_kernel<<<dim3((unsigned)(nBLD / 256)), dim3(256), 0, stream>>>(x, pe, h, hb);

  const dim3 blk256(256), blk128(128);
  const dim3 g_qkv(D_ / 128, MROWS / 128);
  const dim3 g_mlp1(FF_ / 128, MROWS / 128);
  const dim3 g_mlp2(D_ / 128, MROWS / 128);
  const dim3 g_attn(L_ / 64, B_ * H_);
  const dim3 g_ln(B_ * L_);

  for (int l = 0; l < NL_; l++) {
    const size_t wo = (size_t)l * D_ * D_;
    gemm_kernel<0><<<g_qkv, blk256, 0, stream>>>(hb, wq + wo, nullptr, qb, D_, D_);
    gemm_kernel<0><<<g_qkv, blk256, 0, stream>>>(hb, wk + wo, nullptr, kb, D_, D_);
    gemm_kernel<0><<<g_qkv, blk256, 0, stream>>>(hb, wv + wo, nullptr, vb, D_, D_);
    attn_kernel<<<g_attn, blk128, 0, stream>>>(qb, kb, vb, delta);
    ln_kernel<<<g_ln, blk256, 0, stream>>>(h, delta, ln1g + (size_t)l * D_, ln1b + (size_t)l * D_, hb);
    gemm_kernel<1><<<g_mlp1, blk256, 0, stream>>>(hb, w1 + (size_t)l * D_ * FF_, b1 + (size_t)l * FF_, ff1, FF_, D_);
    gemm_kernel<2><<<g_mlp2, blk256, 0, stream>>>(ff1, w2 + (size_t)l * FF_ * D_, b2 + (size_t)l * D_, delta, D_, FF_);
    ln_kernel<<<g_ln, blk256, 0, stream>>>(h, delta, ln2g + (size_t)l * D_, ln2b + (size_t)l * D_, hb);
  }
  hipMemcpyAsync(d_out, h, nBLD * sizeof(float), hipMemcpyDeviceToDevice, stream);
}
